// RotatingCompressiveKVCache_75376676045084
// MI455X (gfx1250) — matrix operand order repaired and checked by reading, not compiled, not run on hardware
//
#include <hip/hip_runtime.h>
#include <hip/hip_bf16.h>

typedef __attribute__((ext_vector_type(16))) __bf16 v16bf;
typedef __attribute__((ext_vector_type(8)))  __bf16 v8bf;
typedef __attribute__((ext_vector_type(8)))  float  v8f;

#define NB    64
#define S2    4096
#define BUFSZ 1024
#define CAND  5120
#define CDIM  64
#define KDIM  128
#define SORTN 8192

// ---------------- CDNA5 async global->LDS staging ----------------------------

static __device__ __forceinline__ void async_b128(unsigned lds_off,
                                                  const __bf16* g) {
  asm volatile("global_load_async_to_lds_b128 %0, %1, off"
               :
               : "v"(lds_off), "v"(g)
               : "memory");
}
static __device__ __forceinline__ void wait_async0() {
  asm volatile("s_wait_asynccnt 0x0" ::: "memory");
}
static __device__ __forceinline__ unsigned lds_off_of(const void* p) {
  return (unsigned)(unsigned long long)p;  // LDS addr = flat addr[31:0]
}
// n must be a multiple of 8*nth so no lane diverges around the async op.
static __device__ __forceinline__ void stage(__bf16* dst, const __bf16* src,
                                             int n, int tid, int nth) {
  const unsigned base = lds_off_of(dst);
  const int chunks = n >> 3;  // 16B chunks
  for (int i = tid; i < chunks; i += nth)
    async_b128(base + i * 16, src + (size_t)i * 8);
}

// ---------------- WMMA helpers (layouts per CDNA5 ISA 7.12.2) ----------------

static __device__ __forceinline__ v8f wmma_bf16(v16bf a, v16bf b, v8f c) {
  return __builtin_amdgcn_wmma_f32_16x16x32_bf16(false, a, false, b, (short)0, c,
                                                 false, false);
}
static __device__ __forceinline__ v8f zero8() { v8f z = {}; return z; }

// A operand 16x32 bf16 from f32 row-major; per-lane row pointer.
// lanes 0-15: K = k0 + {0..7,16..23}; lanes 16-31: K = k0 + {8..15,24..31}
static __device__ __forceinline__ v16bf ldA_f32(const float* rowp, int lane,
                                                int k0) {
  const int kh = (lane & 16) ? 8 : 0;
  const float* p = rowp + k0 + kh;
  v16bf a;
#pragma unroll
  for (int e = 0; e < 8; ++e) a[e] = (__bf16)p[e];
#pragma unroll
  for (int e = 0; e < 8; ++e) a[e + 8] = (__bf16)p[e + 16];
  return a;
}

// Same from bf16 source (global or LDS), 16B-vectorized.
static __device__ __forceinline__ v16bf ldA_bf(const __bf16* rowp, int lane,
                                               int k0) {
  const int kh = (lane & 16) ? 8 : 0;
  const __bf16* p = rowp + k0 + kh;
  v8bf lo = *(const v8bf*)p;
  v8bf hi = *(const v8bf*)(p + 16);
  v16bf a;
#pragma unroll
  for (int e = 0; e < 8; ++e) { a[e] = lo[e]; a[e + 8] = hi[e]; }
  return a;
}

// B operand 32x16 bf16 from row-major weight W[N][K] in LDS, 16B-vectorized.
// lanes 0-15: K = k0 + {0..7,16..23}; lanes 16-31: K = k0 + {8..15,24..31}
static __device__ __forceinline__ v16bf ldB_lds(const __bf16* W, int ldk,
                                                int lane, int n0, int k0) {
  const int n  = lane & 15;
  const int kh = (lane & 16) ? 8 : 0;
  const __bf16* p = W + (size_t)(n0 + n) * ldk + k0 + kh;
  v8bf lo = *(const v8bf*)p;
  v8bf hi = *(const v8bf*)(p + 16);
  v16bf b;
#pragma unroll
  for (int e = 0; e < 8; ++e) { b[e] = lo[e]; b[e + 8] = hi[e]; }
  return b;
}

// D/C 16x16 f32: VGPR g -> row g (lanes<16) / g+8 (lanes>=16), col = lane&15
static __device__ __forceinline__ void stD_f32(float* out, int ld, int lane,
                                               v8f d) {
  const int n  = lane & 15;
  const int rh = (lane & 16) ? 8 : 0;
#pragma unroll
  for (int g = 0; g < 8; ++g) out[(size_t)(g + rh) * ld + n] = d[g];
}
static __device__ __forceinline__ void stD_bf(__bf16* out, int ld, int lane,
                                              v8f d, bool relu) {
  const int n  = lane & 15;
  const int rh = (lane & 16) ? 8 : 0;
#pragma unroll
  for (int g = 0; g < 8; ++g) {
    float x = d[g];
    if (relu && x < 0.f) x = 0.f;
    out[(size_t)(g + rh) * ld + n] = (__bf16)x;
  }
}
static __device__ __forceinline__ v8f bias8(const float* b, int lane, int n0) {
  float v = b[n0 + (lane & 15)];
  v8f c;
#pragma unroll
  for (int g = 0; g < 8; ++g) c[g] = v;
  return c;
}

// fc2 + fc3 tail. h1 [16][128] bf16 / h2 [16][64] bf16: wave-private LDS.
static __device__ __forceinline__ void mlp_tail(__bf16* h1, __bf16* h2,
                                                const __bf16* fc2_s,
                                                const float* fc2_b,
                                                const float* fc3_w,
                                                const float* fc3_b, int lane,
                                                float* imp_out) {
  const __bf16* h1row = h1 + (size_t)(lane & 15) * 128;
  v16bf A[4];
#pragma unroll
  for (int kt = 0; kt < 4; ++kt) A[kt] = ldA_bf(h1row, lane, kt * 32);
#pragma unroll
  for (int nt = 0; nt < 4; ++nt) {
    v8f acc = bias8(fc2_b, lane, nt * 16);
#pragma unroll
    for (int kt = 0; kt < 4; ++kt)
      acc = wmma_bf16(A[kt], ldB_lds(fc2_s, 128, lane, nt * 16, kt * 32), acc);
    stD_bf(h2 + nt * 16, 64, lane, acc, true);
  }
  if (lane < 16) {
    float s = fc3_b[0];
    const __bf16* row = h2 + (size_t)lane * 64;
#pragma unroll 8
    for (int c = 0; c < 64; ++c) s += (float)row[c] * fc3_w[c];
    imp_out[lane] = 1.f / (1.f + __expf(-s));
  }
}

// ---------------- K0: weight prep (bf16 conversion + fused fc1∘reconstruct) --

__global__ __launch_bounds__(256) void prep_kernel(
    const float* wk_p, const float* wk_r, const float* wv_p, const float* wv_r,
    const float* fc1_w, const float* fc2_w, __bf16* wkp_bf, __bf16* wvp_bf,
    __bf16* wkr_bf, __bf16* wvr_bf, __bf16* fc1_bf, __bf16* fc2_bf,
    __bf16* w1ek_bf, __bf16* w1ev_bf) {
  int g = blockIdx.x * 256 + threadIdx.x;
  if (g < 8192) {
    wkp_bf[g] = (__bf16)wk_p[g];
  } else if (g < 16384) {
    int i = g - 8192; wvp_bf[i] = (__bf16)wv_p[i];
  } else if (g < 24576) {
    int i = g - 16384; wkr_bf[i] = (__bf16)wk_r[i];
  } else if (g < 32768) {
    int i = g - 24576; wvr_bf[i] = (__bf16)wv_r[i];
  } else if (g < 65536) {
    int i = g - 32768; fc1_bf[i] = (__bf16)fc1_w[i];
  } else if (g < 73728) {
    int i = g - 65536; fc2_bf[i] = (__bf16)fc2_w[i];
  } else if (g < 81920) {
    int i = g - 73728; int h = i >> 6, c = i & 63;
    float s = 0.f;
    for (int d = 0; d < 128; ++d) s += fc1_w[h * 256 + d] * wk_r[d * 64 + c];
    w1ek_bf[i] = (__bf16)s;
  } else if (g < 90112) {
    int i = g - 81920; int h = i >> 6, c = i & 63;
    float s = 0.f;
    for (int d = 0; d < 128; ++d) s += fc1_w[h * 256 + 128 + d] * wv_r[d * 64 + c];
    w1ev_bf[i] = (__bf16)s;
  }
}

// ---------------- K1: compress keys1/values1 into candidate tail -------------

__global__ __launch_bounds__(256) void compress_old_kernel(
    const float* keys1, const float* values1, const __bf16* wkp_bf,
    const __bf16* wvp_bf, __bf16* cand_k, __bf16* cand_v) {
  __shared__ __align__(16) __bf16 smem[16384];  // wkp 8192 | wvp 8192
  __bf16* wkp_s = smem;
  __bf16* wvp_s = smem + 8192;
  const int tid  = threadIdx.x;
  const int lane = tid & 31;
  const int wv   = tid >> 5;
  stage(wkp_s, wkp_bf, 8192, tid, 256);
  stage(wvp_s, wvp_bf, 8192, tid, 256);
  wait_async0();
  __syncthreads();

  const int gt   = blockIdx.x * 8 + wv;  // 64 tiles/batch
  const int b    = gt >> 6;
  const int trow = (gt & 63) << 4;
  const float* krow = keys1   + ((size_t)b * BUFSZ + trow + (lane & 15)) * KDIM;
  const float* vrow = values1 + ((size_t)b * BUFSZ + trow + (lane & 15)) * KDIM;
  v16bf Ak[4], Av[4];
#pragma unroll
  for (int kt = 0; kt < 4; ++kt) {
    Ak[kt] = ldA_f32(krow, lane, kt * 32);
    Av[kt] = ldA_f32(vrow, lane, kt * 32);
  }
  __bf16* ck = cand_k + ((size_t)b * CAND + S2 + trow) * CDIM;
  __bf16* cv = cand_v + ((size_t)b * CAND + S2 + trow) * CDIM;
#pragma unroll
  for (int nt = 0; nt < 4; ++nt) {
    v8f a = zero8(), c = zero8();
#pragma unroll
    for (int kt = 0; kt < 4; ++kt) {
      a = wmma_bf16(Ak[kt], ldB_lds(wkp_s, 128, lane, nt * 16, kt * 32), a);
      c = wmma_bf16(Av[kt], ldB_lds(wvp_s, 128, lane, nt * 16, kt * 32), c);
    }
    stD_bf(ck + nt * 16, CDIM, lane, a, false);
    stD_bf(cv + nt * 16, CDIM, lane, c, false);
  }
}

// ---------------- K2: new-token importance + compression ---------------------

__global__ __launch_bounds__(256) void new_imp_kernel(
    const float* keys2, const float* values2, const __bf16* wkp_bf,
    const __bf16* wvp_bf, const __bf16* fc1_bf, const __bf16* fc2_bf,
    const float* fc1_b, const float* fc2_b, const float* fc3_w,
    const float* fc3_b, __bf16* cand_k, __bf16* cand_v, float* cand_imp) {
  // wkp 8192 | wvp 8192 | fc1 32768 | fc2 8192 | per-wave h1/h2 8*3072
  __shared__ __align__(16) __bf16 smem[81920];
  __bf16* wkp_s = smem;
  __bf16* wvp_s = smem + 8192;
  __bf16* fc1_s = smem + 16384;
  __bf16* fc2_s = smem + 49152;
  const int tid  = threadIdx.x;
  const int lane = tid & 31;
  const int wv   = tid >> 5;
  __bf16* h1 = smem + 57344 + (size_t)wv * 3072;
  __bf16* h2 = h1 + 2048;

  stage(wkp_s, wkp_bf, 8192, tid, 256);
  stage(wvp_s, wvp_bf, 8192, tid, 256);
  stage(fc1_s, fc1_bf, 32768, tid, 256);
  stage(fc2_s, fc2_bf, 8192, tid, 256);
  wait_async0();
  __syncthreads();

  const int gt   = blockIdx.x * 8 + wv;  // 256 tiles/batch
  const int b    = gt >> 8;
  const int trow = (gt & 255) << 4;

  const float* krow = keys2   + ((size_t)b * S2 + trow + (lane & 15)) * KDIM;
  const float* vrow = values2 + ((size_t)b * S2 + trow + (lane & 15)) * KDIM;
  v16bf Ak[4], Av[4];
#pragma unroll
  for (int kt = 0; kt < 4; ++kt) {
    Ak[kt] = ldA_f32(krow, lane, kt * 32);
    Av[kt] = ldA_f32(vrow, lane, kt * 32);
  }

  // compressed candidates (rows 0..4095)
  __bf16* ck = cand_k + ((size_t)b * CAND + trow) * CDIM;
  __bf16* cv = cand_v + ((size_t)b * CAND + trow) * CDIM;
#pragma unroll
  for (int nt = 0; nt < 4; ++nt) {
    v8f a = zero8(), c = zero8();
#pragma unroll
    for (int kt = 0; kt < 4; ++kt) {
      a = wmma_bf16(Ak[kt], ldB_lds(wkp_s, 128, lane, nt * 16, kt * 32), a);
      c = wmma_bf16(Av[kt], ldB_lds(wvp_s, 128, lane, nt * 16, kt * 32), c);
    }
    stD_bf(ck + nt * 16, CDIM, lane, a, false);
    stD_bf(cv + nt * 16, CDIM, lane, c, false);
  }

  // h1 = relu([K|V] @ fc1^T + b1), 128 outputs
#pragma unroll
  for (int nt = 0; nt < 8; ++nt) {
    v8f acc = bias8(fc1_b, lane, nt * 16);
#pragma unroll
    for (int kt = 0; kt < 4; ++kt)
      acc = wmma_bf16(Ak[kt], ldB_lds(fc1_s, 256, lane, nt * 16, kt * 32), acc);
#pragma unroll
    for (int kt = 0; kt < 4; ++kt)
      acc = wmma_bf16(Av[kt], ldB_lds(fc1_s, 256, lane, nt * 16, 128 + kt * 32),
                      acc);
    stD_bf(h1 + nt * 16, 128, lane, acc, true);
  }
  mlp_tail(h1, h2, fc2_s, fc2_b, fc3_w, fc3_b, lane,
           cand_imp + (size_t)b * CAND + trow);
}

// ---------------- K3: old-entry importance (fused reconstruct∘fc1) -----------

__global__ __launch_bounds__(256) void old_imp_kernel(
    const __bf16* cand_k, const __bf16* cand_v, const __bf16* w1ek_bf,
    const __bf16* w1ev_bf, const __bf16* fc2_bf, const float* fc1_b,
    const float* fc2_b, const float* fc3_w, const float* fc3_b,
    float* cand_imp) {
  // w1ek 8192 | w1ev 8192 | fc2 8192 | per-wave h1/h2 8*3072
  __shared__ __align__(16) __bf16 smem[49152];
  __bf16* w1ek_s = smem;
  __bf16* w1ev_s = smem + 8192;
  __bf16* fc2_s  = smem + 16384;
  const int tid  = threadIdx.x;
  const int lane = tid & 31;
  const int wv   = tid >> 5;
  __bf16* h1 = smem + 24576 + (size_t)wv * 3072;
  __bf16* h2 = h1 + 2048;

  stage(w1ek_s, w1ek_bf, 8192, tid, 256);
  stage(w1ev_s, w1ev_bf, 8192, tid, 256);
  stage(fc2_s, fc2_bf, 8192, tid, 256);
  wait_async0();
  __syncthreads();

  const int gt   = blockIdx.x * 8 + wv;  // 64 tiles/batch
  const int b    = gt >> 6;
  const int trow = (gt & 63) << 4;

  const __bf16* krow =
      cand_k + ((size_t)b * CAND + S2 + trow + (lane & 15)) * CDIM;
  const __bf16* vrow =
      cand_v + ((size_t)b * CAND + S2 + trow + (lane & 15)) * CDIM;
  v16bf Ak[2], Av[2];
#pragma unroll
  for (int kt = 0; kt < 2; ++kt) {
    Ak[kt] = ldA_bf(krow, lane, kt * 32);
    Av[kt] = ldA_bf(vrow, lane, kt * 32);
  }
#pragma unroll
  for (int nt = 0; nt < 8; ++nt) {
    v8f acc = bias8(fc1_b, lane, nt * 16);
#pragma unroll
    for (int kt = 0; kt < 2; ++kt) {
      acc = wmma_bf16(Ak[kt], ldB_lds(w1ek_s, 64, lane, nt * 16, kt * 32), acc);
      acc = wmma_bf16(Av[kt], ldB_lds(w1ev_s, 64, lane, nt * 16, kt * 32), acc);
    }
    stD_bf(h1 + nt * 16, 128, lane, acc, true);
  }
  mlp_tail(h1, h2, fc2_s, fc2_b, fc3_w, fc3_b, lane,
           cand_imp + (size_t)b * CAND + S2 + trow);
}

// ---------------- K4: per-batch top-1024 via LDS bitonic sort ----------------

__global__ __launch_bounds__(1024) void topk_kernel(const float* cand_imp,
                                                    unsigned* top_idx) {
  __shared__ unsigned long long keys[SORTN];
  const int b = blockIdx.x, tid = threadIdx.x;
  for (int i = tid; i < SORTN; i += 1024) {
    unsigned long long kk = 0ull;
    if (i < CAND) {
      unsigned bits = __float_as_uint(cand_imp[(size_t)b * CAND + i]);
      kk = ((unsigned long long)bits << 32) |
           (unsigned long long)(0xFFFFFFFFu - (unsigned)i);
    }
    keys[i] = kk;
  }
  __syncthreads();
  for (int k = 2; k <= SORTN; k <<= 1) {
    for (int j = k >> 1; j > 0; j >>= 1) {
      for (int t = tid; t < SORTN; t += 1024) {
        int ixj = t ^ j;
        if (ixj > t) {
          unsigned long long a = keys[t], c = keys[ixj];
          bool desc = ((t & k) == 0);  // descending overall
          if (desc ? (a < c) : (a > c)) { keys[t] = c; keys[ixj] = a; }
        }
      }
      __syncthreads();
    }
  }
  if (tid < 1024)
    top_idx[(size_t)b * BUFSZ + tid] =
        0xFFFFFFFFu - (unsigned)(keys[tid] & 0xFFFFFFFFull);
}

// ---------------- K5: gather winners + reconstruct to 128-dim ----------------

__global__ __launch_bounds__(256) void gather_out_kernel(
    const __bf16* cand_k, const __bf16* cand_v, const unsigned* top_idx,
    const __bf16* wkr_bf, const __bf16* wvr_bf, float* out_k, float* out_v,
    float* mask) {
  __shared__ __align__(16) __bf16 smem[16384];  // wkr 8192 | wvr 8192
  __bf16* wkr_s = smem;
  __bf16* wvr_s = smem + 8192;
  const int tid  = threadIdx.x;
  const int lane = tid & 31;
  const int wv   = tid >> 5;
  stage(wkr_s, wkr_bf, 8192, tid, 256);
  stage(wvr_s, wvr_bf, 8192, tid, 256);
  wait_async0();
  __syncthreads();

  const int gt   = blockIdx.x * 8 + wv;  // 64 tiles/batch
  const int b    = gt >> 6;
  const int trow = (gt & 63) << 4;

  const unsigned idx = top_idx[(size_t)b * BUFSZ + trow + (lane & 15)];
  const __bf16* krow = cand_k + ((size_t)b * CAND + idx) * CDIM;
  const __bf16* vrow = cand_v + ((size_t)b * CAND + idx) * CDIM;
  v16bf Ak[2], Av[2];
#pragma unroll
  for (int kt = 0; kt < 2; ++kt) {
    Ak[kt] = ldA_bf(krow, lane, kt * 32);
    Av[kt] = ldA_bf(vrow, lane, kt * 32);
  }
  float* ok = out_k + ((size_t)b * BUFSZ + trow) * KDIM;
  float* ov = out_v + ((size_t)b * BUFSZ + trow) * KDIM;
#pragma unroll
  for (int nt = 0; nt < 8; ++nt) {
    v8f a = zero8(), c = zero8();
#pragma unroll
    for (int kt = 0; kt < 2; ++kt) {
      a = wmma_bf16(Ak[kt], ldB_lds(wkr_s, 64, lane, nt * 16, kt * 32), a);
      c = wmma_bf16(Av[kt], ldB_lds(wvr_s, 64, lane, nt * 16, kt * 32), c);
    }
    stD_f32(ok + nt * 16, KDIM, lane, a);
    stD_f32(ov + nt * 16, KDIM, lane, c);
  }
  if (lane < 16) mask[(size_t)b * BUFSZ + trow + lane] = 1.0f;
}

// ---------------- launcher ---------------------------------------------------

extern "C" void kernel_launch(void* const* d_in, const int* in_sizes, int n_in,
                              void* d_out, int out_size, void* d_ws,
                              size_t ws_size, hipStream_t stream) {
  const float* keys1   = (const float*)d_in[0];
  const float* values1 = (const float*)d_in[1];
  const float* keys2   = (const float*)d_in[2];
  const float* values2 = (const float*)d_in[3];
  const float* wk_p    = (const float*)d_in[4];
  const float* wk_r    = (const float*)d_in[5];
  const float* wv_p    = (const float*)d_in[6];
  const float* wv_r    = (const float*)d_in[7];
  const float* fc1_w   = (const float*)d_in[8];
  const float* fc1_b   = (const float*)d_in[9];
  const float* fc2_w   = (const float*)d_in[10];
  const float* fc2_b   = (const float*)d_in[11];
  const float* fc3_w   = (const float*)d_in[12];
  const float* fc3_b   = (const float*)d_in[13];

  char* ws = (char*)d_ws;
  constexpr size_t SZ_CAND   = (size_t)NB * CAND * CDIM * sizeof(__bf16);
  constexpr size_t OFF_CANDK = 0;
  constexpr size_t OFF_CANDV = OFF_CANDK + SZ_CAND;
  constexpr size_t OFF_IMP   = OFF_CANDV + SZ_CAND;
  constexpr size_t OFF_TIDX  = OFF_IMP + (size_t)NB * CAND * sizeof(float);
  constexpr size_t OFF_WKP   = OFF_TIDX + (size_t)NB * BUFSZ * sizeof(unsigned);
  constexpr size_t OFF_WVP   = OFF_WKP + 8192 * sizeof(__bf16);
  constexpr size_t OFF_WKR   = OFF_WVP + 8192 * sizeof(__bf16);
  constexpr size_t OFF_WVR   = OFF_WKR + 8192 * sizeof(__bf16);
  constexpr size_t OFF_FC1   = OFF_WVR + 8192 * sizeof(__bf16);
  constexpr size_t OFF_FC2   = OFF_FC1 + 32768 * sizeof(__bf16);
  constexpr size_t OFF_W1EK  = OFF_FC2 + 8192 * sizeof(__bf16);
  constexpr size_t OFF_W1EV  = OFF_W1EK + 8192 * sizeof(__bf16);

  __bf16* cand_k  = (__bf16*)(ws + OFF_CANDK);
  __bf16* cand_v  = (__bf16*)(ws + OFF_CANDV);
  float*  candimp = (float*)(ws + OFF_IMP);
  unsigned* tidx  = (unsigned*)(ws + OFF_TIDX);
  __bf16* wkp_bf  = (__bf16*)(ws + OFF_WKP);
  __bf16* wvp_bf  = (__bf16*)(ws + OFF_WVP);
  __bf16* wkr_bf  = (__bf16*)(ws + OFF_WKR);
  __bf16* wvr_bf  = (__bf16*)(ws + OFF_WVR);
  __bf16* fc1_bf  = (__bf16*)(ws + OFF_FC1);
  __bf16* fc2_bf  = (__bf16*)(ws + OFF_FC2);
  __bf16* w1ek_bf = (__bf16*)(ws + OFF_W1EK);
  __bf16* w1ev_bf = (__bf16*)(ws + OFF_W1EV);

  float* out_k = (float*)d_out;
  float* out_v = out_k + (size_t)NB * BUFSZ * KDIM;
  float* mask  = out_v + (size_t)NB * BUFSZ * KDIM;

  prep_kernel<<<352, 256, 0, stream>>>(wk_p, wk_r, wv_p, wv_r, fc1_w, fc2_w,
                                       wkp_bf, wvp_bf, wkr_bf, wvr_bf, fc1_bf,
                                       fc2_bf, w1ek_bf, w1ev_bf);
  compress_old_kernel<<<512, 256, 0, stream>>>(keys1, values1, wkp_bf, wvp_bf,
                                               cand_k, cand_v);
  new_imp_kernel<<<2048, 256, 0, stream>>>(keys2, values2, wkp_bf, wvp_bf,
                                           fc1_bf, fc2_bf, fc1_b, fc2_b, fc3_w,
                                           fc3_b, cand_k, cand_v, candimp);
  old_imp_kernel<<<512, 256, 0, stream>>>(cand_k, cand_v, w1ek_bf, w1ev_bf,
                                          fc2_bf, fc1_b, fc2_b, fc3_w, fc3_b,
                                          candimp);
  topk_kernel<<<NB, 1024, 0, stream>>>(candimp, tidx);
  gather_out_kernel<<<512, 256, 0, stream>>>(cand_k, cand_v, tidx, wkr_bf,
                                             wvr_bf, out_k, out_v, mask);
}
